// GraphSAGESOTA_35880156791256
// MI455X (gfx1250) — matrix operand order repaired and checked by reading, not compiled, not run on hardware
//
#include <hip/hip_runtime.h>
#include <hip/hip_bf16.h>

// ---------------------------------------------------------------------------
// GraphSAGE forward on gfx1250 (MI455X).
//  - bf16 WMMA GEMM (v_wmma_f32_16x16x32_bf16), f32 accumulate
//  - weight tile staged to LDS via the Tensor Data Mover (tensor_load_to_lds
//    + s_wait_tensorcnt), 32KB chunks, B fragments then read via ds_load
//  - memory-bound phases (edge scatter / pool) use fp32 atomics
// ---------------------------------------------------------------------------

typedef __attribute__((ext_vector_type(16))) __bf16       v16bf;
typedef __attribute__((ext_vector_type(8)))  float        v8f;
typedef __attribute__((ext_vector_type(4)))  unsigned int v4u;
typedef __attribute__((ext_vector_type(8)))  int          v8i;
typedef __attribute__((ext_vector_type(4)))  int          v4i;

#define HID 128
#define KTOT 256   // concatenated K = 128 (agg) + 128 (root)

#if __has_builtin(__builtin_amdgcn_tensor_load_to_lds) && \
    __has_builtin(__builtin_amdgcn_s_wait_tensorcnt)
#define HAVE_TDM 1
#else
#define HAVE_TDM 0
#endif

// ---------------------------------------------------------------- zero fill
__global__ void k_zero(float* __restrict__ p, long n) {
  long i = (long)blockIdx.x * blockDim.x + threadIdx.x;
  if (i < n) p[i] = 0.0f;
}

// ------------------------------------------------------------- in-degree
__global__ void k_degree(const int* __restrict__ dst, float* __restrict__ deg, int E) {
  int e = blockIdx.x * blockDim.x + threadIdx.x;
  if (e < E) atomicAdd(&deg[dst[e]], 1.0f);
}

// ------------------------------------------------------ nodes per graph
__global__ void k_graph_count(const int* __restrict__ batch, float* __restrict__ cnt, int N) {
  int n = blockIdx.x * blockDim.x + threadIdx.x;
  if (n < N) atomicAdd(&cnt[batch[n]], 1.0f);
}

// -------------------------------------------- pack [Wl | Wr] -> bf16, all layers
__global__ void k_wpack(const float* __restrict__ Wl, const float* __restrict__ Wr,
                        __bf16* __restrict__ Wc, long total) {
  long gid = (long)blockIdx.x * blockDim.x + threadIdx.x;
  if (gid >= total) return;
  long l   = gid / (HID * KTOT);
  long rem = gid % (HID * KTOT);
  long o   = rem / KTOT;
  long k   = rem % KTOT;
  float v = (k < HID) ? Wl[(l * HID + o) * HID + k]
                      : Wr[(l * HID + o) * HID + (k - HID)];
  Wc[gid] = (__bf16)v;
}

// ------------------------------------ edge scatter: agg[dst] += x[src]
__global__ __launch_bounds__(256)
void k_scatter(const int* __restrict__ src, const int* __restrict__ dst,
               const float* __restrict__ x, float* __restrict__ agg, int E) {
  long gid = (long)blockIdx.x * blockDim.x + threadIdx.x;
  int e    = (int)(gid >> 5);
  int lane = (int)(gid & 31);
  if (e >= E) return;
  int s = src[e];
  int d = dst[e];
  const float4 v = *reinterpret_cast<const float4*>(x + (size_t)s * HID + lane * 4);
  float* a = agg + (size_t)d * HID + lane * 4;
  atomicAdd(a + 0, v.x);
  atomicAdd(a + 1, v.y);
  atomicAdd(a + 2, v.z);
  atomicAdd(a + 3, v.w);
}

// --------------------- build bf16 A matrix: A[n] = [agg[n]*inv_deg[n] | x[n]]
__global__ __launch_bounds__(256)
void k_pack(const float* __restrict__ agg, const float* __restrict__ x,
            const float* __restrict__ deg, __bf16* __restrict__ A, int N) {
  long gid = (long)blockIdx.x * blockDim.x + threadIdx.x;
  if (gid >= (long)N * HID) return;
  int n = (int)(gid >> 7);
  int k = (int)(gid & 127);
  float idg = 1.0f / fmaxf(deg[n], 1.0f);
  A[(size_t)n * KTOT + k]       = (__bf16)(agg[gid] * idg);
  A[(size_t)n * KTOT + HID + k] = (__bf16)(x[gid]);
}

// ---------------------------------------------------------------------------
// Fused GEMM + bias + BN partial stats.
//   h[N, 128] = A[N, 256](bf16) @ W[128, 256]^T(bf16) + bias
// Block = 256 threads = 8 waves; block tile = 128 rows x 128 cols.
// Weights staged to LDS in two 32KB K-chunks with the Tensor Data Mover;
// B fragments read from LDS, A fragments streamed from global (L2/HBM).
// 8 K-steps of 32 -> 64 v_wmma_f32_16x16x32_bf16 per wave.
//
// Fragment layouts used here (wave32; contiguous K, same order for A and B):
//   A 16x32 bf16 : lane<16 -> row=lane,    K = kb..kb+15
//                  lane>=16 -> row=lane-16, K = kb+16..kb+31
//   B 32x16 bf16 : lane<16 -> col=lane,    K = kb..kb+15  (contiguous v16bf)
//                  lane>=16 -> col=lane-16, K = kb+16..kb+31
//   C/D 16x16 f32: vgpr r -> row = r + (lane<16 ? 0 : 8), col = lane&15
// ---------------------------------------------------------------------------
__global__ __launch_bounds__(256)
void k_gemm_bn(const __bf16* __restrict__ A, const __bf16* __restrict__ W,
               const float* __restrict__ bias, float* __restrict__ h,
               float* __restrict__ stats, int N) {
  __shared__ float s_sum[HID];
  __shared__ float s_sq[HID];
  __shared__ __align__(32) __bf16 sW[HID * 128];   // 32 KB: one K-chunk of W

  const int tid = threadIdx.x;
  if (tid < HID) { s_sum[tid] = 0.0f; s_sq[tid] = 0.0f; }

  const int wave    = tid >> 5;
  const int lane    = tid & 31;
  const int halfsel = lane >> 4;   // 0: lanes 0-15, 1: lanes 16-31
  const int l16     = lane & 15;
  const int mbase   = blockIdx.x * 128 + wave * 16;

  const int arow  = mbase + l16;                 // A row this lane loads
  const int akoff = halfsel ? 16 : 0;            // A K sub-offset
  const int bkoff = halfsel ? 16 : 0;            // B K sub-offset

  v8f acc[8] = {};
  union AFrag { v16bf v; float4 q[2]; };

  for (int p = 0; p < 2; ++p) {                  // two 32KB weight chunks
    __syncthreads();                             // sW reuse / s_sum init fence
#if HAVE_TDM
    if (tid == 0) {
      // Tensor DMA descriptor (D#): 2D tile, 128 rows x 128 bf16, row
      // stride 256 elements (k-chunked view of W[128][256]).
      const unsigned lds_off = (unsigned)(uintptr_t)(&sW[0]);
      const unsigned long long ga =
          (unsigned long long)(uintptr_t)(W + (size_t)p * 128);
      v4u g0; v8i g1; v4i g2; v4i g3;
      g0[0] = 1u;                                          // count=1, user D#
      g0[1] = lds_off;                                     // lds_addr
      g0[2] = (unsigned)(ga & 0xFFFFFFFFull);              // global_addr lo
      g0[3] = (unsigned)((ga >> 32) & 0x01FFFFFFull)       // global_addr hi
              | (2u << 30);                                // type=2 (image)
      g1[0] = (1 << 16);           // workgroup_mask=0, data_size=1 (2 bytes)
      g1[1] = (128 << 16);         // tensor_dim0[15:0] = 128
      g1[2] = (128 << 16);         // tensor_dim0 hi=0 | tensor_dim1[15:0]=128
      g1[3] = (128 << 16);         // tensor_dim1 hi=0 | tile_dim0 = 128
      g1[4] = 128;                 // tile_dim1 = 128, tile_dim2 = 0
      g1[5] = 256;                 // tensor_dim0_stride = 256 (lo32)
      g1[6] = 0;                   // stride hi / dim1_stride lo
      g1[7] = 0;
      g2[0] = g2[1] = g2[2] = g2[3] = 0;
      g3[0] = g3[1] = g3[2] = g3[3] = 0;
#if defined(__clang_major__) && __clang_major__ >= 23
      v8i gz; for (int i = 0; i < 8; ++i) gz[i] = 0;
      __builtin_amdgcn_tensor_load_to_lds(g0, g1, g2, g3, gz, 0);
#else
      __builtin_amdgcn_tensor_load_to_lds(g0, g1, g2, g3, 0);
#endif
    }
    if (tid < 32) __builtin_amdgcn_s_wait_tensorcnt(0);
#else
    // Fallback: cooperative vectorized copy of the 32KB chunk.
    for (int i = tid; i < HID * 16; i += 256) {
      int col = i >> 4;
      int kv  = (i & 15) * 8;
      *reinterpret_cast<float4*>(&sW[col * 128 + kv]) =
          *reinterpret_cast<const float4*>(W + (size_t)col * KTOT + p * 128 + kv);
    }
#endif
    __syncthreads();

    #pragma unroll
    for (int ksl = 0; ksl < 4; ++ksl) {
      const int kbase = p * 128 + ksl * 32;
      AFrag af;
      if (arow < N) {
        const __bf16* ap = A + (size_t)arow * KTOT + kbase + akoff;
        af.q[0] = *reinterpret_cast<const float4*>(ap);       // K=kb+ak..+7
        af.q[1] = *reinterpret_cast<const float4*>(ap + 8);   // K=kb+ak+8..+15
        __builtin_prefetch(ap + 32, 0, 3);                    // next K-step
      } else {
        af.q[0] = make_float4(0.f, 0.f, 0.f, 0.f);
        af.q[1] = make_float4(0.f, 0.f, 0.f, 0.f);
      }
      #pragma unroll
      for (int t = 0; t < 8; ++t) {
        const int bcol = t * 16 + l16;
        v16bf bf = *reinterpret_cast<const v16bf*>(
            &sW[(size_t)bcol * 128 + ksl * 32 + bkoff]);
        acc[t] = __builtin_amdgcn_wmma_f32_16x16x32_bf16(
            /*neg_a=*/false, af.v, /*neg_b=*/false, bf,
            /*c_mod=*/(short)0, acc[t], /*reuse_a=*/false, /*reuse_b=*/false);
      }
    }
  }

  // Epilogue: bias, store h, per-column BN partial sums (LDS atomics).
  #pragma unroll
  for (int t = 0; t < 8; ++t) {
    const int col = t * 16 + l16;
    const float b = bias[col];
    float psum = 0.0f, psq = 0.0f;
    #pragma unroll
    for (int r = 0; r < 8; ++r) {
      const int row = mbase + r + (halfsel ? 8 : 0);
      const float v = acc[t][r] + b;
      if (row < N) {
        h[(size_t)row * HID + col] = v;
        psum += v;
        psq  += v * v;
      }
    }
    atomicAdd(&s_sum[col], psum);
    atomicAdd(&s_sq[col], psq);
  }
  __syncthreads();
  if (tid < HID) {
    atomicAdd(&stats[tid],       s_sum[tid]);
    atomicAdd(&stats[HID + tid], s_sq[tid]);
  }
}

// ------------------------------------------------- BN: stats -> scale/shift
__global__ void k_bn_finalize(const float* __restrict__ stats,
                              const float* __restrict__ gamma,
                              const float* __restrict__ beta,
                              float* __restrict__ coeff, int N) {
  int c = threadIdx.x;
  if (c >= HID) return;
  float invN = 1.0f / (float)N;
  float mu  = stats[c] * invN;
  float var = fmaxf(stats[HID + c] * invN - mu * mu, 0.0f);
  float s   = rsqrtf(var + 1e-5f) * gamma[c];
  coeff[c]       = s;
  coeff[HID + c] = beta[c] - mu * s;
}

// --------------------------------------------- x = relu(h*scale + shift)
__global__ __launch_bounds__(256)
void k_bn_apply(const float* __restrict__ h, const float* __restrict__ coeff,
                float* __restrict__ xout, long total) {
  long gid = (long)blockIdx.x * blockDim.x + threadIdx.x;
  if (gid >= total) return;
  int k = (int)(gid & 127);
  float v = h[gid] * coeff[k] + coeff[HID + k];
  xout[gid] = fmaxf(v, 0.0f);
}

// ------------------------------------------------ pooled[g] += x[n] (sum)
__global__ __launch_bounds__(256)
void k_pool(const float* __restrict__ x, const int* __restrict__ batch,
            float* __restrict__ pooled, int N) {
  long gid = (long)blockIdx.x * blockDim.x + threadIdx.x;
  int n    = (int)(gid >> 5);
  int lane = (int)(gid & 31);
  if (n >= N) return;
  int g = batch[n];
  const float4 v = *reinterpret_cast<const float4*>(x + (size_t)n * HID + lane * 4);
  float* p = pooled + (size_t)g * HID + lane * 4;
  atomicAdd(p + 0, v.x);
  atomicAdd(p + 1, v.y);
  atomicAdd(p + 2, v.z);
  atomicAdd(p + 3, v.w);
}

// ---------------------------------- classifier: relu(pool @ W1^T+b1) @ W2^T+b2
__global__ __launch_bounds__(128)
void k_classifier(const float* __restrict__ pooled, const float* __restrict__ cnt,
                  const float* __restrict__ W1, const float* __restrict__ b1,
                  const float* __restrict__ W2, const float* __restrict__ b2,
                  float* __restrict__ out) {
  __shared__ float sp[HID];
  __shared__ float sh1[64];
  const int g = blockIdx.x;
  const int tid = threadIdx.x;
  const float inv = 1.0f / fmaxf(cnt[g], 1.0f);
  sp[tid] = pooled[(size_t)g * HID + tid] * inv;
  __syncthreads();
  if (tid < 64) {
    float a = b1[tid];
    const float* wrow = W1 + (size_t)tid * HID;
    #pragma unroll 8
    for (int k = 0; k < HID; ++k) a = fmaf(sp[k], wrow[k], a);
    sh1[tid] = fmaxf(a, 0.0f);
  }
  __syncthreads();
  if (tid < 10) {
    float a = b2[tid];
    const float* wrow = W2 + (size_t)tid * 64;
    #pragma unroll 8
    for (int k = 0; k < 64; ++k) a = fmaf(sh1[k], wrow[k], a);
    out[(size_t)g * 10 + tid] = a;
  }
}

// ---------------------------------------------------------------------------
extern "C" void kernel_launch(void* const* d_in, const int* in_sizes, int n_in,
                              void* d_out, int out_size, void* d_ws, size_t ws_size,
                              hipStream_t stream) {
  const float* x_in  = (const float*)d_in[0];
  const int*   ei    = (const int*)  d_in[1];
  const int*   batch = (const int*)  d_in[2];
  const float* Wl    = (const float*)d_in[3];
  const float* bl    = (const float*)d_in[4];
  const float* Wr    = (const float*)d_in[5];
  const float* gamma = (const float*)d_in[6];
  const float* beta  = (const float*)d_in[7];
  const float* W1    = (const float*)d_in[8];
  const float* b1    = (const float*)d_in[9];
  const float* W2    = (const float*)d_in[10];
  const float* b2    = (const float*)d_in[11];
  float* out = (float*)d_out;

  const int N = in_sizes[0] / HID;          // 50000
  const int E = in_sizes[1] / 2;            // 600000
  const int L = in_sizes[3] / (HID * HID);  // 4
  const int G = out_size / 10;              // 512

  const int* src = ei;
  const int* dst = ei + E;

  // ---- workspace carve-out (256B aligned) ----
  char* base = (char*)d_ws;
  size_t off = 0;
  auto carve = [&](size_t bytes) -> void* {
    void* p = base + off;
    off = (off + bytes + 255) & ~(size_t)255;
    return p;
  };
  float*  xbuf0  = (float*) carve((size_t)N * HID * 4);
  float*  xbuf1  = (float*) carve((size_t)N * HID * 4);
  float*  agg    = (float*) carve((size_t)N * HID * 4);
  float*  hbuf   = (float*) carve((size_t)N * HID * 4);
  __bf16* Abf    = (__bf16*)carve((size_t)N * KTOT * 2);
  __bf16* Wc     = (__bf16*)carve((size_t)L * HID * KTOT * 2);
  float*  deg    = (float*) carve((size_t)N * 4);
  float*  stats  = (float*) carve(2 * HID * 4);
  float*  coeff  = (float*) carve(2 * HID * 4);
  float*  pooled = (float*) carve((size_t)G * HID * 4);
  float*  gcnt   = (float*) carve((size_t)G * 4);
  (void)ws_size; (void)n_in;

  auto cdiv = [](long a, long b) -> int { return (int)((a + b - 1) / b); };
  const long NH = (long)N * HID;

  // ---- one-time per-launch prep ----
  k_zero<<<cdiv(N, 256), 256, 0, stream>>>(deg, N);
  k_degree<<<cdiv(E, 256), 256, 0, stream>>>(dst, deg, E);
  k_zero<<<cdiv(G, 256), 256, 0, stream>>>(gcnt, G);
  k_graph_count<<<cdiv(N, 256), 256, 0, stream>>>(batch, gcnt, N);
  {
    long wtot = (long)L * HID * KTOT;
    k_wpack<<<cdiv(wtot, 256), 256, 0, stream>>>(Wl, Wr, Wc, wtot);
  }
  k_zero<<<cdiv((long)G * HID, 256), 256, 0, stream>>>(pooled, (long)G * HID);

  // ---- layer loop ----
  const float* xcur = x_in;
  float* xping[2] = {xbuf0, xbuf1};
  const int gemm_blocks = cdiv(N, 128);

  for (int l = 0; l < L; ++l) {
    k_zero<<<cdiv(NH, 256), 256, 0, stream>>>(agg, NH);
    k_zero<<<1, 256, 0, stream>>>(stats, 2 * HID);

    k_scatter<<<cdiv((long)E * 32, 256), 256, 0, stream>>>(src, dst, xcur, agg, E);
    k_pack<<<cdiv(NH, 256), 256, 0, stream>>>(agg, xcur, deg, Abf, N);

    k_gemm_bn<<<gemm_blocks, 256, 0, stream>>>(
        Abf, Wc + (size_t)l * HID * KTOT, bl + (size_t)l * HID, hbuf, stats, N);

    k_bn_finalize<<<1, 128, 0, stream>>>(
        stats, gamma + (size_t)l * HID, beta + (size_t)l * HID, coeff, N);
    k_bn_apply<<<cdiv(NH, 256), 256, 0, stream>>>(hbuf, coeff, xping[l & 1], NH);
    xcur = xping[l & 1];
  }

  // ---- pool + classifier ----
  k_pool<<<cdiv((long)N * 32, 256), 256, 0, stream>>>(xcur, batch, pooled, N);
  k_classifier<<<G, 128, 0, stream>>>(pooled, gcnt, W1, b1, W2, b2, out);
}
